// Char2RNN_23545010717003
// MI455X (gfx1250) — matrix operand order repaired and checked by reading, not compiled, not run on hardware
//
#include <hip/hip_runtime.h>
#include <hip/hip_bf16.h>
#include <math.h>

#define Bsz 64
#define Lsz 512
#define Vsz 256
#define Esz 128
#define Hsz 512

typedef __attribute__((ext_vector_type(16))) __bf16 v16bf;
typedef __attribute__((ext_vector_type(8)))  __bf16 v8bf;
typedef __attribute__((ext_vector_type(8)))  float  v8f;

union V16 { v16bf v; v8bf h[2]; };

// ---------------------------------------------------------------------------
// Pack a logical B-operand  B[k][n] = src[n*nstride + k]  into the per-lane
// WMMA 16-bit B layout.  Tile (kt, nt) -> 32 lanes x 16 halfs, stored so a
// wave fetches its whole operand with one contiguous 32-byte load per lane.
// lane: n = nt*16 + (lane&15); hi = lane>>4; element j: k = kt*32 + hi*8 + j (j<8), kt*32 + 16 + hi*8 + (j-8) (j>=8)
// ---------------------------------------------------------------------------
__global__ void pack_b(__bf16* __restrict__ dst, const float* __restrict__ src,
                       int NT, size_t nstride, int total) {
  int gid = blockIdx.x * blockDim.x + threadIdx.x;
  if (gid >= total) return;
  int j    = gid & 15;
  int lane = (gid >> 4) & 31;
  int tile = gid >> 9;
  int nt = tile % NT;
  int kt = tile / NT;
  int n = nt * 16 + (lane & 15);
  int k = kt * 32 + (lane >> 4) * 8 + (j & 7) + (j >> 3) * 16;
  dst[gid] = (__bf16)src[(size_t)n * nstride + (size_t)k];
}

// e stored [L][B][E]:  e[l*B + b][:] = bf16(emb[x[b,l], :])
__global__ void embed_bf16(__bf16* __restrict__ e, const int* __restrict__ x,
                           const float* __restrict__ emb) {
  int gid = blockIdx.x * blockDim.x + threadIdx.x;   // L*B*E threads
  int j   = gid & (Esz - 1);
  int row = gid >> 7;                                // row = l*B + b
  int b   = row & (Bsz - 1);
  int l   = row >> 6;
  int tok = x[b * Lsz + l];
  e[gid] = (__bf16)emb[(size_t)tok * Esz + j];
}

// ---------------------------------------------------------------------------
// xa = e @ diag(A)^T ; xu = e @ U_w^T + bias     (M=L*B, N=H, K=E)
// rows ordered l*B+b everywhere -> fully contiguous stores, and the scan
// reads one contiguous [16][H] slab per step per workgroup.
// ---------------------------------------------------------------------------
__global__ void __launch_bounds__(256)
gemm_xa_xu(const __bf16* __restrict__ e, const __bf16* __restrict__ Apk,
           const __bf16* __restrict__ Upk, const float* __restrict__ bias,
           float* __restrict__ xa, float* __restrict__ xu) {
  int lane = threadIdx.x & 31;
  int wid  = blockIdx.x * 8 + (threadIdx.x >> 5);
  int mt = wid >> 5;                 // 2048 m-tiles
  int nt = wid & 31;                 // 32 n-tiles
  int m0 = mt * 16;
  int col = lane & 15, hi = lane >> 4;
  int n = nt * 16 + col;
  float bv = bias[n];
  v8f accA = {};
  v8f accU;
  for (int r = 0; r < 8; ++r) accU[r] = bv;
  for (int kt = 0; kt < Esz / 32; ++kt) {
    V16 a;
    const __bf16* rowp = e + (size_t)(m0 + col) * Esz + kt * 32 + hi * 8;
    a.h[0] = *(const v8bf*)rowp;
    a.h[1] = *(const v8bf*)(rowp + 16);
    v16bf ba = *(const v16bf*)(Apk + (size_t)(kt * 32 + nt) * 512 + lane * 16);
    v16bf bu = *(const v16bf*)(Upk + (size_t)(kt * 32 + nt) * 512 + lane * 16);
    accA = __builtin_amdgcn_wmma_f32_16x16x32_bf16(false, a.v, false, ba,
                                                   (short)0, accA, false, false);
    accU = __builtin_amdgcn_wmma_f32_16x16x32_bf16(false, a.v, false, bu,
                                                   (short)0, accU, false, false);
  }
  for (int r = 0; r < 8; ++r) {
    size_t idx = (size_t)(m0 + r + 8 * hi) * Hsz + n;   // row = l*B + b
    xa[idx] = accA[r];
    xu[idx] = accU[r];
  }
}

// ---------------------------------------------------------------------------
// Sequential scan: h = tanh(h*xa_t + h @ V_w^T + xu_t)
// one workgroup per 16 batch rows; h kept in LDS (bf16, padded rows);
// 16 waves x 2 N-tiles x 16 K-tiles of v_wmma per step; V_w^T packed + L2-hot.
// Next step's xa/xu slab (2 x 32 KB) is prefetched while WMMAs run.
// hs stored [L][B][H] so per-step writes are one contiguous slab.
// ---------------------------------------------------------------------------
__global__ void __launch_bounds__(512)
rnn_scan(const float* __restrict__ xa, const float* __restrict__ xu,
         const __bf16* __restrict__ Vpk, __bf16* __restrict__ hs,
         float* __restrict__ hT) {
  __shared__ __bf16 hbuf[16][Hsz + 8];      // +8 halfs: 16B-aligned, no bank conflicts
  int tid  = threadIdx.x;
  int lane = tid & 31;
  int w    = tid >> 5;                      // 16 waves
  int b0   = blockIdx.x * 16;
  int col = lane & 15, hi = lane >> 4;

  for (int i = tid; i < 16 * (Hsz + 8); i += blockDim.x)
    ((__bf16*)hbuf)[i] = (__bf16)0.f;
  __syncthreads();

  int nt0 = w * 2;
  for (int t = 0; t < Lsz; ++t) {
    // prefetch next step's elementwise operands (covers 2 x 32 KB exactly)
    if (t + 1 < Lsz) {
      size_t nbase = ((size_t)(t + 1) * Bsz + b0) * Hsz;
      const float* p = (tid < 256 ? xa : xu) + nbase + (size_t)(tid & 255) * 32;
      __builtin_prefetch(p, 0, 0);
    }
    float vals[2][8];
    for (int q = 0; q < 2; ++q) {
      int nt = nt0 + q;
      int n = nt * 16 + col;
      v8f acc = {};
      for (int kt = 0; kt < Hsz / 32; ++kt) {
        V16 a;
        const __bf16* rowp = &hbuf[col][kt * 32 + hi * 8];
        a.h[0] = *(const v8bf*)rowp;                      // ds_load_b128
        a.h[1] = *(const v8bf*)(rowp + 16);
        v16bf bv = *(const v16bf*)(Vpk + (size_t)(kt * 32 + nt) * 512 + lane * 16);
        acc = __builtin_amdgcn_wmma_f32_16x16x32_bf16(false, a.v, false, bv,
                                                      (short)0, acc, false, false);
      }
      for (int r = 0; r < 8; ++r) {
        int m = r + 8 * hi;
        size_t idx = ((size_t)t * Bsz + (b0 + m)) * Hsz + n;
        float hp = (float)hbuf[m][n];
        vals[q][r] = tanhf(acc[r] + hp * xa[idx] + xu[idx]);
      }
    }
    __syncthreads();                        // all WMMA reads of h done
    for (int q = 0; q < 2; ++q) {
      int nt = nt0 + q;
      int n = nt * 16 + col;
      for (int r = 0; r < 8; ++r) {
        int m = r + 8 * hi;
        float vv = vals[q][r];
        hbuf[m][n] = (__bf16)vv;
        hs[((size_t)t * Bsz + (b0 + m)) * Hsz + n] = (__bf16)vv;  // contiguous slab
        if (t == Lsz - 1) hT[(size_t)(b0 + m) * Hsz + n] = vv;
      }
    }
    __syncthreads();                        // h updated for next step
  }
}

// ---------------------------------------------------------------------------
// logits = hs @ dec_w^T + dec_b    (M=L*B rows in t*B+b order, N=V, K=H)
// A-tiles read contiguous hs rows; only the final store scatters to [B][L][V].
// ---------------------------------------------------------------------------
__global__ void __launch_bounds__(256)
gemm_logits(const __bf16* __restrict__ hs, const __bf16* __restrict__ Dpk,
            const float* __restrict__ dec_b, float* __restrict__ out) {
  int lane = threadIdx.x & 31;
  int wid  = blockIdx.x * 8 + (threadIdx.x >> 5);
  int mt = wid >> 4;                 // 2048 m-tiles
  int nt = wid & 15;                 // 16 n-tiles
  int m0 = mt * 16;
  int col = lane & 15, hi = lane >> 4;
  int n = nt * 16 + col;
  float bv = dec_b[n];
  v8f acc;
  for (int r = 0; r < 8; ++r) acc[r] = bv;
  for (int kt = 0; kt < Hsz / 32; ++kt) {
    V16 a;
    const __bf16* rowp = hs + (size_t)(m0 + col) * Hsz + kt * 32 + hi * 8;
    a.h[0] = *(const v8bf*)rowp;
    a.h[1] = *(const v8bf*)(rowp + 16);
    v16bf bb = *(const v16bf*)(Dpk + (size_t)(kt * 16 + nt) * 512 + lane * 16);
    acc = __builtin_amdgcn_wmma_f32_16x16x32_bf16(false, a.v, false, bb,
                                                  (short)0, acc, false, false);
  }
  for (int r = 0; r < 8; ++r) {
    int row = m0 + r + 8 * hi;       // row = t*B + b
    int b = row & (Bsz - 1);
    int t = row >> 6;
    out[((size_t)b * Lsz + t) * Vsz + n] = acc[r];
  }
}

extern "C" void kernel_launch(void* const* d_in, const int* in_sizes, int n_in,
                              void* d_out, int out_size, void* d_ws, size_t ws_size,
                              hipStream_t stream) {
  (void)in_sizes; (void)n_in; (void)out_size; (void)ws_size;
  const int*   x     = (const int*)  d_in[0];
  const float* emb   = (const float*)d_in[1];
  const float* A     = (const float*)d_in[2];
  const float* U_w   = (const float*)d_in[3];
  const float* V_w   = (const float*)d_in[4];
  const float* bias  = (const float*)d_in[5];
  const float* dec_w = (const float*)d_in[6];
  const float* dec_b = (const float*)d_in[7];
  float* out = (float*)d_out;

  char* ws = (char*)d_ws;
  size_t off = 0;
  auto carve = [&](size_t bytes) {
    char* p = ws + off;
    off = (off + bytes + 255) & ~(size_t)255;
    return p;
  };
  __bf16* e_bf  = (__bf16*)carve((size_t)Lsz * Bsz * Esz * 2);  //  8 MB  [L][B][E]
  __bf16* Apk   = (__bf16*)carve((size_t)Esz * Hsz * 2);        // 128 KB
  __bf16* Upk   = (__bf16*)carve((size_t)Esz * Hsz * 2);        // 128 KB
  __bf16* Vpk   = (__bf16*)carve((size_t)Hsz * Hsz * 2);        // 512 KB
  __bf16* Dpk   = (__bf16*)carve((size_t)Hsz * Vsz * 2);        // 256 KB
  float*  xa    = (float*) carve((size_t)Lsz * Bsz * Hsz * 4);  // 64 MB  [L][B][H]
  float*  xu    = (float*) carve((size_t)Lsz * Bsz * Hsz * 4);  // 64 MB  [L][B][H]
  __bf16* hsbuf = (__bf16*)carve((size_t)Lsz * Bsz * Hsz * 2);  // 32 MB  [L][B][H]

  // B-operand packing:  B[k][n] = src[n*nstride + k]
  pack_b<<<(Esz * Hsz) / 256, 256, 0, stream>>>(Apk, A,     32, (size_t)(Hsz + 1) * Esz, Esz * Hsz); // diag(A)
  pack_b<<<(Esz * Hsz) / 256, 256, 0, stream>>>(Upk, U_w,   32, (size_t)Esz,             Esz * Hsz);
  pack_b<<<(Hsz * Hsz) / 256, 256, 0, stream>>>(Vpk, V_w,   32, (size_t)Hsz,             Hsz * Hsz);
  pack_b<<<(Hsz * Vsz) / 256, 256, 0, stream>>>(Dpk, dec_w, 16, (size_t)Hsz,             Hsz * Vsz);

  embed_bf16<<<(Lsz * Bsz * Esz) / 256, 256, 0, stream>>>(e_bf, x, emb);

  gemm_xa_xu<<<(2048 * 32) / 8, 256, 0, stream>>>(e_bf, Apk, Upk, bias, xa, xu);

  rnn_scan<<<Bsz / 16, 512, 0, stream>>>(xa, xu, Vpk, hsbuf,
                                         out + (size_t)Bsz * Lsz * Vsz);

  gemm_logits<<<(2048 * 16) / 8, 256, 0, stream>>>(hsbuf, Dpk, dec_b, out);
}
